// AttnRegGNN_41704132444692
// MI455X (gfx1250) — matrix operand order repaired and checked by reading, not compiled, not run on hardware
//
#include <hip/hip_runtime.h>

// Problem constants (from reference)
#define N_NODES 40000
#define N_EDGES 640000
#define HID     128
#define HEADS   4
#define ED      16
#define LAYERS  3
#define OUT_DIM 7
#define JK      (HID * LAYERS)   // 384

typedef __attribute__((ext_vector_type(16))) __bf16 v16bf;
typedef __attribute__((ext_vector_type(8)))  __bf16 v8bf;
typedef __attribute__((ext_vector_type(8)))  float  v8f;

__device__ __forceinline__ unsigned short f32_to_bf16_rne(float f) {
    unsigned u = __float_as_uint(f);
    unsigned r = 0x7FFFu + ((u >> 16) & 1u);
    return (unsigned short)((u + r) >> 16);
}

__device__ __forceinline__ void atomic_max_float(float* addr, float val) {
    // IEEE trick: positive floats ordered as ints, negative as reversed uints.
    if (val >= 0.0f) atomicMax((int*)addr, __float_as_int(val));
    else             atomicMin((unsigned int*)addr, (unsigned int)__float_as_int(val));
}

// ---------------------------------------------------------------------------
// Utility kernels
// ---------------------------------------------------------------------------
__global__ void fill_kernel(float* __restrict__ p, float v, int n) {
    int i = blockIdx.x * blockDim.x + threadIdx.x;
    if (i < n) p[i] = v;
}

__global__ void cvt_bf16_kernel(const float* __restrict__ s, unsigned short* __restrict__ d, int n) {
    int i = blockIdx.x * blockDim.x + threadIdx.x;
    if (i < n) d[i] = f32_to_bf16_rne(s[i]);
}

// W is K x Nc row-major (f32); Wt becomes Nc x K row-major (bf16)
__global__ void transpose_cvt_kernel(const float* __restrict__ W, unsigned short* __restrict__ Wt,
                                     int K, int Nc) {
    int i = blockIdx.x * blockDim.x + threadIdx.x;
    if (i >= K * Nc) return;
    int k = i / Nc, n = i % Nc;
    Wt[(size_t)n * K + k] = f32_to_bf16_rne(W[i]);
}

// ---------------------------------------------------------------------------
// WMMA GEMM: Y[M x 128] = Xb[M x 128](bf16) @ W(128x128) + bias, W given as
// Wt[n][k] (bf16, transposed). 8 waves/block, one 16-row strip per wave.
// ---------------------------------------------------------------------------
__global__ __launch_bounds__(256)
void gemm128_wmma_kernel(const unsigned short* __restrict__ Xb,
                         const unsigned short* __restrict__ Wt,
                         const float* __restrict__ bias,
                         float* __restrict__ Y, int M) {
    const int K = 128, Nc = 128;
    int wave = threadIdx.x >> 5;
    int lane = threadIdx.x & 31;
    int rowTile = blockIdx.x * 8 + wave;
    if (rowTile * 16 >= M) return;            // wave-uniform exit: EXEC stays all-ones
    int row0 = rowTile * 16;
    int r    = row0 + (lane & 15);
    int half = lane >> 4;
    int c    = lane & 15;

    // Preload A fragments (K/32 = 4). ISA 16-bit A 16x32 layout:
    // elems 0..7 : k = 32kt + (half?8:0) + j ; elems 8..15: k = 32kt + 16 + (half?8:0) + j
    v16bf afrag[4];
#pragma unroll
    for (int kt = 0; kt < 4; ++kt) {
        const unsigned short* p = Xb + (size_t)r * K + kt * 32 + (half ? 8 : 0);
        v8bf lo = *(const v8bf*)p;
        v8bf hi = *(const v8bf*)(p + 16);
        v16bf a;
#pragma unroll
        for (int j = 0; j < 8; ++j) { a[j] = lo[j]; a[8 + j] = hi[j]; }
        afrag[kt] = a;
    }

#pragma unroll
    for (int nt = 0; nt < Nc / 16; ++nt) {
        int n0 = nt * 16;
        float bv = bias[n0 + c];
        v8f acc;
#pragma unroll
        for (int j = 0; j < 8; ++j) acc[j] = bv;
#pragma unroll
        for (int kt = 0; kt < 4; ++kt) {
            // B 32x16: lane holds col n0+c, k in same order as A: 32kt + (half?8:0) + j, then +16
            const unsigned short* p = Wt + (size_t)(n0 + c) * K + kt * 32 + (half ? 8 : 0);
            v8bf blo = *(const v8bf*)p;
            v8bf bhi = *(const v8bf*)(p + 16);
            v16bf b;
#pragma unroll
            for (int j = 0; j < 8; ++j) { b[j] = blo[j]; b[8 + j] = bhi[j]; }
            acc = __builtin_amdgcn_wmma_f32_16x16x32_bf16(false, afrag[kt], false, b,
                                                          (short)0, acc, false, false);
        }
#pragma unroll
        for (int j = 0; j < 8; ++j) {
            int rr = row0 + j + (half ? 8 : 0);
            Y[(size_t)rr * Nc + n0 + c] = acc[j];
        }
    }
}

// ---------------------------------------------------------------------------
// Edge pass 1: alpha[e][h] = <q[dst], k[src]+e_proj>/sqrt(32); atomic max per dst.
// One wave per edge; lane owns channel h*32+lane of each head.
// ---------------------------------------------------------------------------
__global__ __launch_bounds__(256)
void edge_alpha_kernel(const float* __restrict__ q, const float* __restrict__ k,
                       const float* __restrict__ ea, const float* __restrict__ We,
                       const long long* __restrict__ src, const long long* __restrict__ dst,
                       float* __restrict__ alpha, float* __restrict__ amax) {
    int wave = threadIdx.x >> 5;
    int lane = threadIdx.x & 31;
    long long e = (long long)blockIdx.x * 8 + wave;
    if (e >= N_EDGES) return;
    int s = (int)src[e], d = (int)dst[e];

    float eav[ED];
#pragma unroll
    for (int j = 0; j < ED; ++j) eav[j] = ea[e * ED + j];

    float p[HEADS];
#pragma unroll
    for (int h = 0; h < HEADS; ++h) {
        int ch = h * 32 + lane;
        float ev = 0.f;
#pragma unroll
        for (int j = 0; j < ED; ++j) ev += eav[j] * We[j * HID + ch];
        float kj = k[(size_t)s * HID + ch] + ev;
        p[h] = q[(size_t)d * HID + ch] * kj;
    }
#pragma unroll
    for (int m = 16; m >= 1; m >>= 1)
#pragma unroll
        for (int h = 0; h < HEADS; ++h) p[h] += __shfl_xor(p[h], m, 32);

    const float scale = 0.17677669529663687f; // 1/sqrt(32)
    if (lane == 0) {
#pragma unroll
        for (int h = 0; h < HEADS; ++h) {
            float a = p[h] * scale;
            alpha[e * HEADS + h] = a;
            atomic_max_float(&amax[(size_t)d * HEADS + h], a);
        }
    }
}

// Edge pass 2: ex = exp(alpha - amax[dst]); denom[dst] += ex (thread per edge*head)
__global__ __launch_bounds__(256)
void edge_exp_kernel(float* __restrict__ alpha, const float* __restrict__ amax,
                     float* __restrict__ denom, const long long* __restrict__ dst) {
    long long t = (long long)blockIdx.x * blockDim.x + threadIdx.x;
    if (t >= (long long)N_EDGES * HEADS) return;
    long long e = t >> 2;
    int h = (int)(t & 3);
    int d = (int)dst[e];
    float ex = __expf(alpha[t] - amax[(size_t)d * HEADS + h]);
    alpha[t] = ex;
    atomicAdd(&denom[(size_t)d * HEADS + h], ex);
}

// Edge pass 3: out[dst] += (ex/denom) * (v[src] + e_proj); wave per edge
__global__ __launch_bounds__(256)
void edge_agg_kernel(const float* __restrict__ v, const float* __restrict__ ea,
                     const float* __restrict__ We, const float* __restrict__ ex,
                     const float* __restrict__ denom,
                     const long long* __restrict__ src, const long long* __restrict__ dst,
                     float* __restrict__ out) {
    int wave = threadIdx.x >> 5;
    int lane = threadIdx.x & 31;
    long long e = (long long)blockIdx.x * 8 + wave;
    if (e >= N_EDGES) return;
    int s = (int)src[e], d = (int)dst[e];

    float w[HEADS];
#pragma unroll
    for (int h = 0; h < HEADS; ++h)
        w[h] = ex[e * HEADS + h] / (denom[(size_t)d * HEADS + h] + 1e-16f);

    float eav[ED];
#pragma unroll
    for (int j = 0; j < ED; ++j) eav[j] = ea[e * ED + j];

#pragma unroll
    for (int h = 0; h < HEADS; ++h) {
        int ch = h * 32 + lane;
        float ev = 0.f;
#pragma unroll
        for (int j = 0; j < ED; ++j) ev += eav[j] * We[j * HID + ch];
        float vj = v[(size_t)s * HID + ch] + ev;
        atomicAdd(&out[(size_t)d * HID + ch], w[h] * vj);
    }
}

// Beta gating: comb = beta*skip + (1-beta)*out, beta = sigmoid([out,skip,out-skip]@Wbeta)
__global__ __launch_bounds__(256)
void combine_kernel(const float* __restrict__ out, const float* __restrict__ skip,
                    const float* __restrict__ Wbeta, float* __restrict__ comb) {
    int wave = threadIdx.x >> 5;
    int lane = threadIdx.x & 31;
    int node = blockIdx.x * 8 + wave;
    if (node >= N_NODES) return;
    float s = 0.f;
#pragma unroll
    for (int i = 0; i < 4; ++i) {
        int c = i * 32 + lane;
        float o  = out[(size_t)node * HID + c];
        float sk = skip[(size_t)node * HID + c];
        s += o * Wbeta[c] + sk * Wbeta[HID + c] + (o - sk) * Wbeta[2 * HID + c];
    }
#pragma unroll
    for (int m = 16; m >= 1; m >>= 1) s += __shfl_xor(s, m, 32);
    float beta = 1.f / (1.f + __expf(-s));
#pragma unroll
    for (int i = 0; i < 4; ++i) {
        int c = i * 32 + lane;
        float o  = out[(size_t)node * HID + c];
        float sk = skip[(size_t)node * HID + c];
        comb[(size_t)node * HID + c] = beta * sk + (1.f - beta) * o;
    }
}

// GraphNorm column reductions
__global__ __launch_bounds__(128)
void colsum_kernel(const float* __restrict__ x, float* __restrict__ sums) {
    int c = threadIdx.x;
    float acc = 0.f;
    for (int r = blockIdx.x; r < N_NODES; r += gridDim.x) acc += x[(size_t)r * HID + c];
    atomicAdd(&sums[c], acc);
}

__global__ __launch_bounds__(128)
void colsumsq_kernel(const float* __restrict__ x, const float* __restrict__ mean,
                     const float* __restrict__ ms, float* __restrict__ sums) {
    int c = threadIdx.x;
    float m = ms[c] * mean[c];
    float acc = 0.f;
    for (int r = blockIdx.x; r < N_NODES; r += gridDim.x) {
        float t = x[(size_t)r * HID + c] - m;
        acc += t * t;
    }
    atomicAdd(&sums[c], acc);
}

__global__ void mean_finalize_kernel(const float* __restrict__ sums, float* __restrict__ mean) {
    int c = threadIdx.x;
    mean[c] = sums[c] * (1.0f / N_NODES);
}

__global__ void inv_finalize_kernel(const float* __restrict__ varsum, float* __restrict__ inv) {
    int c = threadIdx.x;
    inv[c] = rsqrtf(varsum[c] * (1.0f / N_NODES) + 1e-5f);
}

// normalize + relu + residual; emit f32 h, bf16 h (next layer GEMM), bf16 JK slot
__global__ __launch_bounds__(256)
void apply_kernel(const float* __restrict__ comb, const float* __restrict__ mean,
                  const float* __restrict__ inv, const float* __restrict__ gnw,
                  const float* __restrict__ gnb, const float* __restrict__ ms,
                  const float* __restrict__ hin, float* __restrict__ hout,
                  unsigned short* __restrict__ hbf, unsigned short* __restrict__ hcat,
                  int layer) {
    int idx = blockIdx.x * blockDim.x + threadIdx.x;
    if (idx >= N_NODES * HID) return;
    int c = idx & (HID - 1);
    size_t n = (size_t)idx >> 7;
    float t = comb[idx] - ms[c] * mean[c];
    float y = gnw[c] * t * inv[c] + gnb[c];
    y = fmaxf(y, 0.f);
    float hn = y + hin[idx];
    hout[idx] = hn;
    unsigned short b = f32_to_bf16_rne(hn);
    hbf[idx] = b;
    hcat[n * JK + (size_t)layer * HID + c] = b;
}

// ---------------------------------------------------------------------------
// Final fused: h1 = relu(hcat(bf16, Nx384) @ W1t + b1) via WMMA into per-wave
// LDS, then 128->7 GEMV with W2/b2, split-store (N x 6 | N x 1).
// ---------------------------------------------------------------------------
__global__ __launch_bounds__(256)
void final_wmma_kernel(const unsigned short* __restrict__ Xb,   // N x 384 bf16
                       const unsigned short* __restrict__ W1t,  // 128 x 384 bf16
                       const float* __restrict__ b1,
                       const float* __restrict__ W2,            // 128 x 7 f32
                       const float* __restrict__ b2,
                       float* __restrict__ outp, int M) {
    const int K = JK, Nc = HID;
    __shared__ float h1s[8 * 16 * HID];     // 64 KB: wave-private 16x128 tiles

    int wave = threadIdx.x >> 5;
    int lane = threadIdx.x & 31;
    int rowTile = blockIdx.x * 8 + wave;
    if (rowTile * 16 >= M) return;          // wave-uniform exit
    int row0 = rowTile * 16;
    int r    = row0 + (lane & 15);
    int half = lane >> 4;
    int c    = lane & 15;

    v16bf afrag[K / 32];                    // 12 fragments
#pragma unroll
    for (int kt = 0; kt < K / 32; ++kt) {
        const unsigned short* p = Xb + (size_t)r * K + kt * 32 + (half ? 8 : 0);
        v8bf lo = *(const v8bf*)p;
        v8bf hi = *(const v8bf*)(p + 16);
        v16bf a;
#pragma unroll
        for (int j = 0; j < 8; ++j) { a[j] = lo[j]; a[8 + j] = hi[j]; }
        afrag[kt] = a;
    }

#pragma unroll
    for (int nt = 0; nt < Nc / 16; ++nt) {
        int n0 = nt * 16;
        float bv = b1[n0 + c];
        v8f acc;
#pragma unroll
        for (int j = 0; j < 8; ++j) acc[j] = bv;
#pragma unroll
        for (int kt = 0; kt < K / 32; ++kt) {
            const unsigned short* p = W1t + (size_t)(n0 + c) * K + kt * 32 + (half ? 8 : 0);
            v8bf blo = *(const v8bf*)p;
            v8bf bhi = *(const v8bf*)(p + 16);
            v16bf b;
#pragma unroll
            for (int j = 0; j < 8; ++j) { b[j] = blo[j]; b[8 + j] = bhi[j]; }
            acc = __builtin_amdgcn_wmma_f32_16x16x32_bf16(false, afrag[kt], false, b,
                                                          (short)0, acc, false, false);
        }
#pragma unroll
        for (int j = 0; j < 8; ++j) {
            int rloc = j + (half ? 8 : 0);
            h1s[(wave * 16 + rloc) * HID + n0 + c] = fmaxf(acc[j], 0.f);
        }
    }

    // second stage (wave-private LDS; in-wave DS ordering, no barrier needed):
    // lanes 0-15: rows 0-15, outputs 0..3; lanes 16-31: rows 0-15, outputs 4..6
    int rloc   = lane & 15;
    int obase  = half ? 4 : 0;
    int ocount = half ? 3 : 4;
    float acc7[4];
#pragma unroll
    for (int oo = 0; oo < 4; ++oo) acc7[oo] = (oo < ocount) ? b2[obase + oo] : 0.f;
    const float* hrow = &h1s[(wave * 16 + rloc) * HID];
#pragma unroll 8
    for (int kk = 0; kk < HID; ++kk) {
        float hv = hrow[kk];
#pragma unroll
        for (int oo = 0; oo < 4; ++oo)
            if (oo < ocount) acc7[oo] += hv * W2[kk * OUT_DIM + obase + oo];
    }
    int n = row0 + rloc;
#pragma unroll
    for (int oo = 0; oo < 4; ++oo) {
        if (oo >= ocount) continue;
        int o = obase + oo;
        if (o < 6) outp[(size_t)n * 6 + o] = acc7[oo];
        else       outp[(size_t)M * 6 + n] = acc7[oo];
    }
}

// ---------------------------------------------------------------------------
// Host-side orchestration
// ---------------------------------------------------------------------------
extern "C" void kernel_launch(void* const* d_in, const int* in_sizes, int n_in,
                              void* d_out, int out_size, void* d_ws, size_t ws_size,
                              hipStream_t stream) {
    (void)in_sizes; (void)n_in; (void)out_size; (void)ws_size;
    const float* x      = (const float*)d_in[0];
    const float* eattr  = (const float*)d_in[1];
    const float* Wq     = (const float*)d_in[2];
    const float* bq     = (const float*)d_in[3];
    const float* Wk     = (const float*)d_in[4];
    const float* bk     = (const float*)d_in[5];
    const float* Wv     = (const float*)d_in[6];
    const float* bv     = (const float*)d_in[7];
    const float* We     = (const float*)d_in[8];
    const float* Wskip  = (const float*)d_in[9];
    const float* bskip  = (const float*)d_in[10];
    const float* Wbeta  = (const float*)d_in[11];
    const float* gnw    = (const float*)d_in[12];
    const float* gnb    = (const float*)d_in[13];
    const float* gnms   = (const float*)d_in[14];
    const float* W1     = (const float*)d_in[15];
    const float* b1     = (const float*)d_in[16];
    const float* W2     = (const float*)d_in[17];
    const float* b2     = (const float*)d_in[18];
    const long long* ei = (const long long*)d_in[19];
    const long long* src = ei;
    const long long* dst = ei + N_EDGES;
    float* outp = (float*)d_out;

    // --- workspace carving (~176 MB) ---
    char* p = (char*)d_ws;
    auto alloc = [&](size_t bytes) -> void* {
        void* r = (void*)p;
        p += (bytes + 255) & ~(size_t)255;
        return r;
    };
    unsigned short* h_bf   = (unsigned short*)alloc((size_t)N_NODES * HID * 2);
    unsigned short* hcat   = (unsigned short*)alloc((size_t)N_NODES * JK * 2);
    float* qbuf   = (float*)alloc((size_t)N_NODES * HID * 4);   // reused as comb
    float* kbuf   = (float*)alloc((size_t)N_NODES * HID * 4);
    float* vbuf   = (float*)alloc((size_t)N_NODES * HID * 4);
    float* sbuf   = (float*)alloc((size_t)N_NODES * HID * 4);
    float* obuf   = (float*)alloc((size_t)N_NODES * HID * 4);
    float* hbuf   = (float*)alloc((size_t)N_NODES * HID * 4);
    float* alphab = (float*)alloc((size_t)N_EDGES * HEADS * 4);
    float* amaxb  = (float*)alloc((size_t)N_NODES * HEADS * 4);
    float* denomb = (float*)alloc((size_t)N_NODES * HEADS * 4);
    float* sums   = (float*)alloc(HID * 4);
    float* meanb  = (float*)alloc(HID * 4);
    float* varsum = (float*)alloc(HID * 4);
    float* invb   = (float*)alloc(HID * 4);
    unsigned short* wt_all = (unsigned short*)alloc((size_t)LAYERS * 4 * HID * HID * 2);
    unsigned short* w1t    = (unsigned short*)alloc((size_t)HID * JK * 2);

    const int TPB = 256;
    const int gemmBlocks = (N_NODES / 16 + 7) / 8;          // 313
    const int edgeWaveBlocks = (N_EDGES + 7) / 8;           // 80000
    const int nodeWaveBlocks = (N_NODES + 7) / 8;           // 5000
    const int elemBlocks = (N_NODES * HID + TPB - 1) / TPB; // 20000

    // --- weight prep: transpose + bf16 ---
    for (int l = 0; l < LAYERS; ++l) {
        const float* Ws[4] = { Wq + (size_t)l * HID * HID, Wk + (size_t)l * HID * HID,
                               Wv + (size_t)l * HID * HID, Wskip + (size_t)l * HID * HID };
        for (int m = 0; m < 4; ++m) {
            unsigned short* dstw = wt_all + ((size_t)l * 4 + m) * HID * HID;
            transpose_cvt_kernel<<<(HID * HID + TPB - 1) / TPB, TPB, 0, stream>>>(Ws[m], dstw, HID, HID);
        }
    }
    transpose_cvt_kernel<<<(JK * HID + TPB - 1) / TPB, TPB, 0, stream>>>(W1, w1t, JK, HID);
    cvt_bf16_kernel<<<elemBlocks, TPB, 0, stream>>>(x, h_bf, N_NODES * HID);

    const float* hin = x;
    for (int l = 0; l < LAYERS; ++l) {
        const unsigned short* wq_t = wt_all + ((size_t)l * 4 + 0) * HID * HID;
        const unsigned short* wk_t = wt_all + ((size_t)l * 4 + 1) * HID * HID;
        const unsigned short* wv_t = wt_all + ((size_t)l * 4 + 2) * HID * HID;
        const unsigned short* ws_t = wt_all + ((size_t)l * 4 + 3) * HID * HID;
        const float* We_l = We + (size_t)l * ED * HID;

        // node projections (WMMA)
        gemm128_wmma_kernel<<<gemmBlocks, TPB, 0, stream>>>(h_bf, wq_t, bq + (size_t)l * HID, qbuf, N_NODES);
        gemm128_wmma_kernel<<<gemmBlocks, TPB, 0, stream>>>(h_bf, wk_t, bk + (size_t)l * HID, kbuf, N_NODES);
        gemm128_wmma_kernel<<<gemmBlocks, TPB, 0, stream>>>(h_bf, wv_t, bv + (size_t)l * HID, vbuf, N_NODES);
        gemm128_wmma_kernel<<<gemmBlocks, TPB, 0, stream>>>(h_bf, ws_t, bskip + (size_t)l * HID, sbuf, N_NODES);

        // segment softmax init
        fill_kernel<<<(N_NODES * HEADS + TPB - 1) / TPB, TPB, 0, stream>>>(amaxb, -INFINITY, N_NODES * HEADS);
        hipMemsetAsync(denomb, 0, (size_t)N_NODES * HEADS * 4, stream);
        hipMemsetAsync(obuf,   0, (size_t)N_NODES * HID * 4, stream);

        edge_alpha_kernel<<<edgeWaveBlocks, TPB, 0, stream>>>(qbuf, kbuf, eattr, We_l, src, dst, alphab, amaxb);
        edge_exp_kernel<<<(N_EDGES * HEADS + TPB - 1) / TPB, TPB, 0, stream>>>(alphab, amaxb, denomb, dst);
        edge_agg_kernel<<<edgeWaveBlocks, TPB, 0, stream>>>(vbuf, eattr, We_l, alphab, denomb, src, dst, obuf);

        // beta gate (comb reuses qbuf)
        combine_kernel<<<nodeWaveBlocks, TPB, 0, stream>>>(obuf, sbuf, Wbeta + (size_t)l * 3 * HID, qbuf);

        // GraphNorm
        hipMemsetAsync(sums,   0, HID * 4, stream);
        hipMemsetAsync(varsum, 0, HID * 4, stream);
        colsum_kernel<<<256, HID, 0, stream>>>(qbuf, sums);
        mean_finalize_kernel<<<1, HID, 0, stream>>>(sums, meanb);
        colsumsq_kernel<<<256, HID, 0, stream>>>(qbuf, meanb, gnms + (size_t)l * HID, varsum);
        inv_finalize_kernel<<<1, HID, 0, stream>>>(varsum, invb);
        apply_kernel<<<elemBlocks, TPB, 0, stream>>>(qbuf, meanb, invb,
                                                     gnw + (size_t)l * HID, gnb + (size_t)l * HID,
                                                     gnms + (size_t)l * HID, hin, hbuf, h_bf, hcat, l);
        hin = hbuf;
    }

    // JK MLP (WMMA) + 128->7 head, split store
    final_wmma_kernel<<<gemmBlocks, TPB, 0, stream>>>(hcat, w1t, b1, W2, b2, outp, N_NODES);
}
